// HeteroGNNLayer_38611755991310
// MI455X (gfx1250) — matrix operand order repaired and checked by reading, not compiled, not run on hardware
//
#include <hip/hip_runtime.h>

typedef __attribute__((ext_vector_type(16))) _Float16 v16h;
typedef __attribute__((ext_vector_type(8)))  _Float16 v8h;
typedef __attribute__((ext_vector_type(8)))  float    v8f;

static constexpr int KD = 128;   // feature dim D
static constexpr int NH = 4;     // heads
static constexpr int XSTR = 136; // LDS stride (halfs) for X tile  (272B: 16B-aligned, bank-skewed)
static constexpr int WSTR = 136; // LDS stride (halfs) for W^T
static constexpr int HSTR = 132; // LDS stride (floats) for C tile (528B: 16B-aligned, bank-skewed)
static constexpr int N_CELL = 100000;
static constexpr int N_IO   = 2000;
static constexpr int E_CC   = 600000;
static constexpr int E_CIO  = 50000;
static constexpr int E_IOC  = 50000;
// NOTE: all row counts (100000, 2000) are multiples of 16 -> no tail guards needed.

// ---------------- float <-> ordered-uint encoding for atomic max ----------------
__device__ __forceinline__ unsigned fenc(float f) {
  unsigned u = __float_as_uint(f);
  return (u >> 31) ? ~u : (u | 0x80000000u);
}
__device__ __forceinline__ float fdec(unsigned e) {
  return (e & 0x80000000u) ? __uint_as_float(e & 0x7fffffffu) : __uint_as_float(~e);
}

// ---------------- WMMA inner engine: one 16x16 tile of X(16x128) @ W(128x128) ----------------
// xs : LDS, 16 rows x XSTR halfs.  wst: LDS, W transposed: wst[n*WSTR + k].
// Lane mapping (wave32): m = lane&15, g = lane>>4, ncol = wave*16 + m.
__device__ __forceinline__ v8f wmma_tile_128(const _Float16* xs, const _Float16* wst,
                                             int m, int g, int ncol) {
  v8f c = {};
#pragma unroll
  for (int kb = 0; kb < 4; ++kb) {
    const int k0 = kb * 32;
    // A fragment (16x32 f16): halves 0-7 -> K = k0 + g*8 + i ; halves 8-15 -> K = k0 + 16 + g*8 + i
    v8h alo = *(const v8h*)(xs + m * XSTR + k0 + g * 8);
    v8h ahi = *(const v8h*)(xs + m * XSTR + k0 + 16 + g * 8);
    // B fragment (32x16 f16): halves 0-7 -> K = k0 + g*8 + i ; halves 8-15 -> K = k0 + 16 + g*8 + i, column ncol
    v8h blo = *(const v8h*)(wst + ncol * WSTR + k0 + g * 8);
    v8h bhi = *(const v8h*)(wst + ncol * WSTR + k0 + 16 + g * 8);
    v16h a, b;
#pragma unroll
    for (int i = 0; i < 8; ++i) { a[i] = alo[i]; a[i + 8] = ahi[i]; b[i] = blo[i]; b[i + 8] = bhi[i]; }
    c = __builtin_amdgcn_wmma_f32_16x16x32_f16(false, a, false, b, (short)0, c, false, false);
  }
  return c;
}

// ---------------- fused projection GEMM + per-head attention logits ----------------
// Hout[r,:] = X[r,:] @ W ; als[r,h] = sum_c Hout[r,h*32+c]*a_s[h*32+c] ; same for ald.
// Any of Hout/als_out/ald_out may be null. Requires nrows % 16 == 0.
__global__ __launch_bounds__(256) void gat_gemm_kernel(
    const float* __restrict__ X, const float* __restrict__ W,
    const float* __restrict__ a_s, const float* __restrict__ a_d,
    float* __restrict__ Hout, float* __restrict__ als_out, float* __restrict__ ald_out,
    int nrows) {
  __shared__ alignas(16) _Float16 wst[KD * WSTR];   // ~34.8 KB, W transposed (bank-skewed)
  __shared__ alignas(16) _Float16 xs[16 * XSTR];    // ~4.3 KB
  __shared__ alignas(16) float hrow[16 * HSTR];     // ~8.4 KB, C tile
  __shared__ float ash[KD], adh[KD];

  const int tid  = threadIdx.x;
  const int lane = tid & 31;
  const int wave = tid >> 5;
  const int row0 = blockIdx.x * 16;
  (void)nrows;

  // stage W^T (coalesced global reads; ~2-way LDS write conflicts from skew)
  for (int i = tid; i < KD * KD; i += 256) {
    int k = i >> 7, n = i & 127;
    wst[n * WSTR + k] = (_Float16)W[i];
  }
  // stage X tile (coalesced, conflict-free)
  for (int i = tid; i < 16 * KD; i += 256) {
    int m = i >> 7, k = i & 127;
    xs[m * XSTR + k] = (_Float16)X[(size_t)(row0 + m) * KD + k];
  }
  if (tid < KD) { ash[tid] = a_s[tid]; adh[tid] = a_d[tid]; }
  __syncthreads();

  const int m = lane & 15, g = lane >> 4;
  const int ncol = wave * 16 + m;
  v8f c = wmma_tile_128(xs, wst, m, g, ncol);

  // stage C tile to LDS (skewed stride: the two lane-groups land in disjoint banks)
#pragma unroll
  for (int j = 0; j < 8; ++j) hrow[(j + 8 * g) * HSTR + ncol] = c[j];
  __syncthreads();

  // coalesced vectorized writeback of H
  if (Hout) {
    for (int i = tid; i < 16 * 32; i += 256) {
      const int mrow = i >> 5, c4 = (i & 31) << 2;
      *(float4*)&Hout[(size_t)(row0 + mrow) * KD + c4] = *(const float4*)&hrow[mrow * HSTR + c4];
    }
  }
  // attention logits: one thread per (row, head)
  if (tid < 16 * NH) {
    const int mrow = tid >> 2, h = tid & 3;
    const float* hr = &hrow[mrow * HSTR + h * 32];
    float ss = 0.f, sd = 0.f;
#pragma unroll
    for (int k = 0; k < 32; ++k) {
      const float v = hr[k];
      ss += v * ash[h * 32 + k];
      sd += v * adh[h * 32 + k];
    }
    if (als_out) als_out[(size_t)(row0 + mrow) * NH + h] = ss;
    if (ald_out) ald_out[(size_t)(row0 + mrow) * NH + h] = sd;
  }
}

// ---------------- edge pass 1: leaky-relu logits + segment max (encoded atomic max) ----------------
__global__ void edge_logit_kernel(const int* __restrict__ ei, int E,
                                  const float* __restrict__ als, const float* __restrict__ ald,
                                  float* __restrict__ ebuf, unsigned* __restrict__ menc) {
  int e = blockIdx.x * blockDim.x + threadIdx.x;
  if (e >= E) return;
  const int s = ei[e], d = ei[E + e];
#pragma unroll
  for (int h = 0; h < NH; ++h) {
    float v = als[s * NH + h] + ald[d * NH + h];
    v = v > 0.f ? v : 0.2f * v;
    ebuf[(size_t)e * NH + h] = v;
    atomicMax(&menc[d * NH + h], fenc(v));
  }
}

// ---------------- edge pass 2: exp(e - m[d]) + segment sum ----------------
__global__ void edge_exp_kernel(const int* __restrict__ ei, int E,
                                float* __restrict__ ebuf, const unsigned* __restrict__ menc,
                                float* __restrict__ den) {
  int e = blockIdx.x * blockDim.x + threadIdx.x;
  if (e >= E) return;
  const int d = ei[E + e];
#pragma unroll
  for (int h = 0; h < NH; ++h) {
    float mx = fdec(menc[d * NH + h]);
    float x = __expf(ebuf[(size_t)e * NH + h] - mx);
    ebuf[(size_t)e * NH + h] = x;
    atomicAdd(&den[d * NH + h], x);
  }
}

// ---------------- edge pass 3: alpha-weighted message scatter (128 lanes per edge) ----------------
__global__ void edge_scatter_kernel(const int* __restrict__ ei, int E,
                                    const float* __restrict__ ebuf, const float* __restrict__ den,
                                    const float* __restrict__ Hs, float* __restrict__ acc) {
  long long t = (long long)blockIdx.x * blockDim.x + threadIdx.x;
  if (t >= (long long)E * KD) return;
  const int e = (int)(t >> 7);
  const int ch = (int)(t & 127);
  const int s = ei[e], d = ei[E + e];
  const int h = ch >> 5;
  const float alpha = ebuf[(size_t)e * NH + h] / (den[d * NH + h] + 1e-16f);
  atomicAdd(&acc[(size_t)d * KD + ch], alpha * Hs[(size_t)s * KD + ch]);
}

// ---------------- fused transform: h = acc@Wt + bt ; LN ; acc + relu(...) ----------------
// Requires nrows % 16 == 0.
__global__ __launch_bounds__(256) void transform_kernel(
    const float* __restrict__ acc, const float* __restrict__ Wt, const float* __restrict__ bt,
    const float* __restrict__ gma, const float* __restrict__ bet,
    float* __restrict__ out, int nrows) {
  __shared__ alignas(16) _Float16 wst[KD * WSTR];
  __shared__ alignas(16) _Float16 xs[16 * XSTR];
  __shared__ alignas(16) float hrow[16 * HSTR];
  __shared__ float mu_sh[16], rs_sh[16];

  const int tid  = threadIdx.x;
  const int lane = tid & 31;
  const int wave = tid >> 5;
  const int row0 = blockIdx.x * 16;
  (void)nrows;

  for (int i = tid; i < KD * KD; i += 256) {
    int k = i >> 7, n = i & 127;
    wst[n * WSTR + k] = (_Float16)Wt[i];
  }
  for (int i = tid; i < 16 * KD; i += 256) {
    int m = i >> 7, k = i & 127;
    xs[m * XSTR + k] = (_Float16)acc[(size_t)(row0 + m) * KD + k];
  }
  __syncthreads();

  const int m = lane & 15, g = lane >> 4;
  const int ncol = wave * 16 + m;
  v8f c = wmma_tile_128(xs, wst, m, g, ncol);

  const float btv = bt[ncol];
#pragma unroll
  for (int j = 0; j < 8; ++j) hrow[(j + 8 * g) * HSTR + ncol] = c[j] + btv;
  __syncthreads();

  if (tid < 16) {
    float s = 0.f, s2 = 0.f;
    for (int k = 0; k < KD; ++k) { float v = hrow[tid * HSTR + k]; s += v; s2 += v * v; }
    const float mu = s * (1.0f / KD);
    const float var = s2 * (1.0f / KD) - mu * mu;
    mu_sh[tid] = mu;
    rs_sh[tid] = rsqrtf(var + 1e-5f);
  }
  __syncthreads();

  // vectorized LN + residual ReLU, coalesced
  for (int i = tid; i < 16 * 32; i += 256) {
    const int mrow = i >> 5, c4 = (i & 31) << 2;
    const float mu = mu_sh[mrow], rs = rs_sh[mrow];
    const float4 hv = *(const float4*)&hrow[mrow * HSTR + c4];
    const float4 av = *(const float4*)&acc[(size_t)(row0 + mrow) * KD + c4];
    const float4 gv = *(const float4*)&gma[c4];
    const float4 bv = *(const float4*)&bet[c4];
    float4 o;
    float h0 = (hv.x - mu) * rs * gv.x + bv.x; o.x = av.x + (h0 > 0.f ? h0 : 0.f);
    float h1 = (hv.y - mu) * rs * gv.y + bv.y; o.y = av.y + (h1 > 0.f ? h1 : 0.f);
    float h2 = (hv.z - mu) * rs * gv.z + bv.z; o.z = av.z + (h2 > 0.f ? h2 : 0.f);
    float h3 = (hv.w - mu) * rs * gv.w + bv.w; o.w = av.w + (h3 > 0.f ? h3 : 0.f);
    *(float4*)&out[(size_t)(row0 + mrow) * KD + c4] = o;
  }
}

// ---------------- init helpers ----------------
__global__ void init_acc_kernel(float* __restrict__ acc, const float* __restrict__ b1,
                                const float* __restrict__ b2, long long total) {
  long long t = (long long)blockIdx.x * blockDim.x + threadIdx.x;
  if (t >= total) return;
  const int ch = (int)(t & 127);
  acc[t] = b1[ch] + (b2 ? b2[ch] : 0.f);
}
__global__ void init_u32_kernel(unsigned* __restrict__ p, unsigned v, int n) {
  int t = blockIdx.x * blockDim.x + threadIdx.x;
  if (t < n) p[t] = v;
}
__global__ void init_f32_kernel(float* __restrict__ p, float v, int n) {
  int t = blockIdx.x * blockDim.x + threadIdx.x;
  if (t < n) p[t] = v;
}

static inline int cdiv_ll(long long a, long long b) { return (int)((a + b - 1) / b); }

extern "C" void kernel_launch(void* const* d_in, const int* in_sizes, int n_in,
                              void* d_out, int out_size, void* d_ws, size_t ws_size,
                              hipStream_t stream) {
  (void)in_sizes; (void)n_in; (void)out_size; (void)ws_size;

  const float* x_cell = (const float*)d_in[0];
  const float* x_io   = (const float*)d_in[1];
  const int*   ei_cc  = (const int*)d_in[2];
  const int*   ei_cio = (const int*)d_in[3];
  const int*   ei_ioc = (const int*)d_in[4];
  const float* W_cc   = (const float*)d_in[5];
  const float* as_cc  = (const float*)d_in[6];
  const float* ad_cc  = (const float*)d_in[7];
  const float* b_cc   = (const float*)d_in[8];
  const float* W_cio  = (const float*)d_in[9];
  const float* as_cio = (const float*)d_in[10];
  const float* ad_cio = (const float*)d_in[11];
  const float* b_cio  = (const float*)d_in[12];
  const float* W_ioc  = (const float*)d_in[13];
  const float* as_ioc = (const float*)d_in[14];
  const float* ad_ioc = (const float*)d_in[15];
  const float* b_ioc  = (const float*)d_in[16];
  const float* Wt_cell = (const float*)d_in[17];
  const float* bt_cell = (const float*)d_in[18];
  const float* g_cell  = (const float*)d_in[19];
  const float* be_cell = (const float*)d_in[20];
  const float* Wt_io   = (const float*)d_in[21];
  const float* bt_io   = (const float*)d_in[22];
  const float* g_io    = (const float*)d_in[23];
  const float* be_io   = (const float*)d_in[24];

  float* out_cell = (float*)d_out;
  float* out_io   = out_cell + (size_t)N_CELL * KD;

  // workspace layout (floats): ~119 MB total, fits the 192 MB L2
  float* ws = (float*)d_ws;
  size_t off = 0;
  float* Hbuf     = ws + off; off += (size_t)N_CELL * KD;   // projected src features (reused per relation)
  float* acc_cell = ws + off; off += (size_t)N_CELL * KD;
  float* acc_io   = ws + off; off += (size_t)N_IO * KD;
  float* ALS      = ws + off; off += (size_t)N_CELL * NH;
  float* ALD      = ws + off; off += (size_t)N_CELL * NH;
  unsigned* Menc  = (unsigned*)(ws + off); off += (size_t)N_CELL * NH;
  float* Den      = ws + off; off += (size_t)N_CELL * NH;
  float* Ebuf     = ws + off; off += (size_t)E_CC * NH;

  const dim3 B(256);
  const unsigned NEG_INF_ENC = 0x007FFFFFu;   // fenc(-inf)

  // accumulators start at the summed per-relation biases (HeteroConv aggr='sum')
  init_acc_kernel<<<cdiv_ll((long long)N_CELL * KD, 256), B, 0, stream>>>(acc_cell, b_cc, b_ioc, (long long)N_CELL * KD);
  init_acc_kernel<<<cdiv_ll((long long)N_IO * KD, 256), B, 0, stream>>>(acc_io, b_cio, nullptr, (long long)N_IO * KD);

  // ======== relation cc: cell -> cell (src == dst, shared projection) ========
  gat_gemm_kernel<<<cdiv_ll(N_CELL, 16), B, 0, stream>>>(x_cell, W_cc, as_cc, ad_cc, Hbuf, ALS, ALD, N_CELL);
  init_u32_kernel<<<cdiv_ll(N_CELL * NH, 256), B, 0, stream>>>(Menc, NEG_INF_ENC, N_CELL * NH);
  init_f32_kernel<<<cdiv_ll(N_CELL * NH, 256), B, 0, stream>>>(Den, 0.f, N_CELL * NH);
  edge_logit_kernel<<<cdiv_ll(E_CC, 256), B, 0, stream>>>(ei_cc, E_CC, ALS, ALD, Ebuf, Menc);
  edge_exp_kernel<<<cdiv_ll(E_CC, 256), B, 0, stream>>>(ei_cc, E_CC, Ebuf, Menc, Den);
  edge_scatter_kernel<<<cdiv_ll((long long)E_CC * KD, 256), B, 0, stream>>>(ei_cc, E_CC, Ebuf, Den, Hbuf, acc_cell);

  // ======== relation ioc: io -> cell ========
  gat_gemm_kernel<<<cdiv_ll(N_IO, 16), B, 0, stream>>>(x_io, W_ioc, as_ioc, ad_ioc, Hbuf, ALS, nullptr, N_IO);
  gat_gemm_kernel<<<cdiv_ll(N_CELL, 16), B, 0, stream>>>(x_cell, W_ioc, as_ioc, ad_ioc, nullptr, nullptr, ALD, N_CELL);
  init_u32_kernel<<<cdiv_ll(N_CELL * NH, 256), B, 0, stream>>>(Menc, NEG_INF_ENC, N_CELL * NH);
  init_f32_kernel<<<cdiv_ll(N_CELL * NH, 256), B, 0, stream>>>(Den, 0.f, N_CELL * NH);
  edge_logit_kernel<<<cdiv_ll(E_IOC, 256), B, 0, stream>>>(ei_ioc, E_IOC, ALS, ALD, Ebuf, Menc);
  edge_exp_kernel<<<cdiv_ll(E_IOC, 256), B, 0, stream>>>(ei_ioc, E_IOC, Ebuf, Menc, Den);
  edge_scatter_kernel<<<cdiv_ll((long long)E_IOC * KD, 256), B, 0, stream>>>(ei_ioc, E_IOC, Ebuf, Den, Hbuf, acc_cell);

  // ======== relation cio: cell -> io ========
  gat_gemm_kernel<<<cdiv_ll(N_CELL, 16), B, 0, stream>>>(x_cell, W_cio, as_cio, ad_cio, Hbuf, ALS, nullptr, N_CELL);
  gat_gemm_kernel<<<cdiv_ll(N_IO, 16), B, 0, stream>>>(x_io, W_cio, as_cio, ad_cio, nullptr, nullptr, ALD, N_IO);
  init_u32_kernel<<<cdiv_ll(N_IO * NH, 256), B, 0, stream>>>(Menc, NEG_INF_ENC, N_IO * NH);
  init_f32_kernel<<<cdiv_ll(N_IO * NH, 256), B, 0, stream>>>(Den, 0.f, N_IO * NH);
  edge_logit_kernel<<<cdiv_ll(E_CIO, 256), B, 0, stream>>>(ei_cio, E_CIO, ALS, ALD, Ebuf, Menc);
  edge_exp_kernel<<<cdiv_ll(E_CIO, 256), B, 0, stream>>>(ei_cio, E_CIO, Ebuf, Menc, Den);
  edge_scatter_kernel<<<cdiv_ll((long long)E_CIO * KD, 256), B, 0, stream>>>(ei_cio, E_CIO, Ebuf, Den, Hbuf, acc_io);

  // ======== per-type transform (WMMA GEMM + LayerNorm + residual ReLU) ========
  transform_kernel<<<cdiv_ll(N_CELL, 16), B, 0, stream>>>(acc_cell, Wt_cell, bt_cell, g_cell, be_cell, out_cell, N_CELL);
  transform_kernel<<<cdiv_ll(N_IO, 16), B, 0, stream>>>(acc_io, Wt_io, bt_io, g_io, be_io, out_io, N_IO);
}
